// QAttention_23819888624285
// MI455X (gfx1250) — compile-verified, matrix operand order checked by reading, not run on hardware
//
#include <hip/hip_runtime.h>
#include <hip/hip_bf16.h>

typedef __attribute__((ext_vector_type(16))) __bf16 v16bf;
typedef __attribute__((ext_vector_type(8)))  float  v8f;

#define EPS_Q 0.01f

__device__ __forceinline__ float pow2ceil(float m) {
  m = fmaxf(m, EPS_Q);
  return exp2f(ceilf(log2f(m)));
}

// ---------------------------------------------------------------------------
// Per-column power-of-2 max-abs scale: s1[n] = 2^ceil(log2(max(eps, max_m|x|)))
// Generic strided/batched: element (m,n) of batch z at
//   X[(z/innerN)*zOut + (z%innerN)*zIn + m*rowStride + n*colStride]
// ---------------------------------------------------------------------------
__global__ void colmax_pow2_kernel(const float* __restrict__ X,
                                   float* __restrict__ s1,
                                   int M, int N,
                                   long long rowStride, long long colStride,
                                   int innerN, long long zOut, long long zIn)
{
  int n = blockIdx.x * blockDim.x + threadIdx.x;
  int z = blockIdx.y;
  if (n >= N) return;
  long long base = (long long)(z / innerN) * zOut + (long long)(z % innerN) * zIn;
  const float* x = X + base + (long long)n * colStride;
  float m = 0.f;
  for (int i = 0; i < M; ++i) m = fmaxf(m, fabsf(x[(long long)i * rowStride]));
  s1[(long long)z * N + n] = pow2ceil(m);
}

// ---------------------------------------------------------------------------
// Row pass: s2 = pow2ceil(max_n |x/s1|); q = clamp(rint(x/(s1*s2)*128));
// out = s1*q*s2/128 stored as bf16 (exact). One block per row.
// ---------------------------------------------------------------------------
__global__ void rowdq_kernel(const float* __restrict__ X,
                             const float* __restrict__ s1,
                             __hip_bfloat16* __restrict__ Out,
                             int N,
                             long long rowStride, long long colStride,
                             int innerN, long long zOut, long long zIn,
                             long long outZStride, int ldOut, int transposeOut)
{
  __shared__ float red[9];
  int m = blockIdx.x;
  int z = blockIdx.y;
  long long base = (long long)(z / innerN) * zOut + (long long)(z % innerN) * zIn
                 + (long long)m * rowStride;
  const float* s1z = s1 + (long long)z * N;

  float mx = 0.f;
  for (int j = threadIdx.x; j < N; j += blockDim.x)
    mx = fmaxf(mx, fabsf(X[base + (long long)j * colStride]) / s1z[j]);
  for (int off = 16; off > 0; off >>= 1) mx = fmaxf(mx, __shfl_xor(mx, off, 32));
  if ((threadIdx.x & 31) == 0) red[threadIdx.x >> 5] = mx;
  __syncthreads();
  if (threadIdx.x == 0) {
    float v = red[0];
    for (int w = 1; w < (int)(blockDim.x >> 5); ++w) v = fmaxf(v, red[w]);
    red[8] = pow2ceil(v);
  }
  __syncthreads();
  float s2 = red[8];
  float qscale = 128.f / s2;
  float rscale = s2 * (1.f / 128.f);
  __hip_bfloat16* outz = Out + (long long)z * outZStride;
  for (int j = threadIdx.x; j < N; j += blockDim.x) {
    float v = X[base + (long long)j * colStride];
    float q = rintf(v / s1z[j] * qscale);        // v_rndne: round-half-even == jnp.round
    q = fminf(fmaxf(q, -128.f), 127.f);
    float d = s1z[j] * q * rscale;
    long long oidx = transposeOut ? (long long)j * ldOut + m
                                  : (long long)m * ldOut + j;
    outz[oidx] = __float2bfloat16(d);
  }
}

// ---------------------------------------------------------------------------
// Causal softmax over row i of the [S,S] score matrix (in place, f32).
// ---------------------------------------------------------------------------
__global__ void softmax_causal_kernel(float* __restrict__ AW, int S, float scale)
{
  __shared__ float red[9];
  int i = blockIdx.x;
  long long z = blockIdx.y;
  float* row = AW + (z * (long long)S + i) * (long long)S;
  int L = i + 1;

  float mx = -3.4e38f;
  for (int j = threadIdx.x; j < L; j += blockDim.x) mx = fmaxf(mx, row[j] * scale);
  for (int off = 16; off > 0; off >>= 1) mx = fmaxf(mx, __shfl_xor(mx, off, 32));
  if ((threadIdx.x & 31) == 0) red[threadIdx.x >> 5] = mx;
  __syncthreads();
  if (threadIdx.x == 0) {
    float v = red[0];
    for (int w = 1; w < 8; ++w) v = fmaxf(v, red[w]);
    red[8] = v;
  }
  __syncthreads();
  mx = red[8];

  float sum = 0.f;
  for (int j = threadIdx.x; j < L; j += blockDim.x) sum += expf(row[j] * scale - mx);
  for (int off = 16; off > 0; off >>= 1) sum += __shfl_xor(sum, off, 32);
  if ((threadIdx.x & 31) == 0) red[threadIdx.x >> 5] = sum;
  __syncthreads();
  if (threadIdx.x == 0) {
    float v = 0.f;
    for (int w = 0; w < 8; ++w) v += red[w];
    red[8] = v;
  }
  __syncthreads();
  float inv = 1.f / red[8];

  for (int j = threadIdx.x; j < L; j += blockDim.x)
    row[j] = expf(row[j] * scale - mx) * inv;
  for (int j = L + threadIdx.x; j < S; j += blockDim.x)
    row[j] = 0.f;
}

// ---------------------------------------------------------------------------
// bf16 WMMA GEMM: C[M,N](f32) = A[M,K] @ Bt[N,K]^T (+ bias[N])
// Block = 64x64x64 tile, 128 threads = 4 wave32, each wave a 32x32 subtile
// (2x2 of v_wmma_f32_16x16x32_bf16).
// Tiles staged with CDNA5 GLOBAL_LOAD_ASYNC_TO_LDS_B128 (ASYNCcnt path,
// bypasses VGPRs), LDS rows padded to 72 bf16 to spread ds_load_b128 banks.
// ---------------------------------------------------------------------------
#define GBM 64
#define GBN 64
#define GBK 64
#define LPAD 72

// Low 32 bits of an LDS-aperture flat address are the LDS byte address.
__device__ __forceinline__ uint32_t lds_off(const void* p) {
  return (uint32_t)(uintptr_t)p;
}

__device__ __forceinline__ void async_load_b128(uint32_t lds_addr, const void* gaddr) {
  asm volatile("global_load_async_to_lds_b128 %0, %1, off"
               :: "v"(lds_addr), "v"(gaddr)
               : "memory");
}

__device__ __forceinline__ void wait_async_all() {
  asm volatile("s_wait_asynccnt 0" ::: "memory");
}

__device__ __forceinline__ v16bf lds_frag(const __hip_bfloat16* base, int row, int kk, int lane) {
  // A 16x32 bf16 layout: lane holds row = base+(lane&15), K = kk + 16*(lane>>4) .. +15
  // (B 32x16 mirrors this with N in place of M since Bt is stored N-major.)
  const __hip_bfloat16* p = base + (row + (lane & 15)) * LPAD + kk + ((lane >> 4) << 4);
  union { v16bf v; uint4 u[2]; } r;
  r.u[0] = *(const uint4*)p;
  r.u[1] = *(const uint4*)(p + 8);
  return r.v;
}

__global__ void __launch_bounds__(128)
gemm_bf16_wmma_kernel(const __hip_bfloat16* __restrict__ A,
                      const __hip_bfloat16* __restrict__ Bt,
                      float* __restrict__ C,
                      const float* __restrict__ bias,
                      int K, int lda, int ldb, int ldc,
                      long long aZ, long long bZ,
                      int cInnerN, long long cZOuter, long long cZInner)
{
  __shared__ __align__(16) __hip_bfloat16 lA[GBM * LPAD];
  __shared__ __align__(16) __hip_bfloat16 lB[GBN * LPAD];

  int tid = threadIdx.x;
  int lane = tid & 31;
  int wave = tid >> 5;
  int wm = (wave >> 1) << 5;   // 0 or 32
  int wn = (wave & 1) << 5;    // 0 or 32
  long long z = blockIdx.z;

  const __hip_bfloat16* Ab = A + z * aZ + (long long)blockIdx.y * GBM * lda;
  const __hip_bfloat16* Bb = Bt + z * bZ + (long long)blockIdx.x * GBN * ldb;

  v8f acc00 = {}, acc01 = {}, acc10 = {}, acc11 = {};

  for (int k0 = 0; k0 < K; k0 += GBK) {
    // Stage 64x64 bf16 tiles of A and Bt: async DMA straight into LDS.
    // 512 16B chunks per matrix / 128 threads = 4 per thread per matrix.
#pragma unroll
    for (int it = 0; it < 4; ++it) {
      int t = tid + (it << 7);        // 0..511
      int r = t >> 3;                  // row 0..63
      int seg = (t & 7) << 3;          // 8-elem (16B) chunk within the 64-wide row
      async_load_b128(lds_off(&lA[r * LPAD + seg]), Ab + (long long)r * lda + k0 + seg);
      async_load_b128(lds_off(&lB[r * LPAD + seg]), Bb + (long long)r * ldb + k0 + seg);
    }
    wait_async_all();
    __syncthreads();
#pragma unroll
    for (int kk = 0; kk < GBK; kk += 32) {
      v16bf a0 = lds_frag(lA, wm, kk, lane);
      v16bf a1 = lds_frag(lA, wm + 16, kk, lane);
      v16bf b0 = lds_frag(lB, wn, kk, lane);
      v16bf b1 = lds_frag(lB, wn + 16, kk, lane);
      acc00 = __builtin_amdgcn_wmma_f32_16x16x32_bf16(false, a0, false, b0, (short)0, acc00, false, false);
      acc01 = __builtin_amdgcn_wmma_f32_16x16x32_bf16(false, a0, false, b1, (short)0, acc01, false, false);
      acc10 = __builtin_amdgcn_wmma_f32_16x16x32_bf16(false, a1, false, b0, (short)0, acc10, false, false);
      acc11 = __builtin_amdgcn_wmma_f32_16x16x32_bf16(false, a1, false, b1, (short)0, acc11, false, false);
    }
    __syncthreads();
  }

  long long cz = (long long)(blockIdx.z / cInnerN) * cZOuter
               + (long long)(blockIdx.z % cInnerN) * cZInner;
  float* Cb = C + cz + (long long)(blockIdx.y * GBM) * ldc + blockIdx.x * GBN;
  int mrow = (lane >> 4) << 3;  // 0 or 8  (C layout: VGPR r -> M = r or 8+r)
  int ncol = lane & 15;

#pragma unroll
  for (int mt = 0; mt < 2; ++mt) {
#pragma unroll
    for (int nt = 0; nt < 2; ++nt) {
      v8f a = (mt == 0) ? (nt == 0 ? acc00 : acc01) : (nt == 0 ? acc10 : acc11);
      int n = wn + nt * 16 + ncol;
      float bv = bias ? bias[blockIdx.x * GBN + n] : 0.f;
#pragma unroll
      for (int r = 0; r < 8; ++r) {
        int mm = wm + mt * 16 + mrow + r;
        Cb[(long long)mm * ldc + n] = a[r] + bv;
      }
    }
  }
}

// ---------------------------------------------------------------------------
extern "C" void kernel_launch(void* const* d_in, const int* in_sizes, int n_in,
                              void* d_out, int out_size, void* d_ws, size_t ws_size,
                              hipStream_t stream) {
  (void)in_sizes; (void)n_in; (void)out_size; (void)ws_size;
  const float* hidden = (const float*)d_in[0];
  const float* w_attn = (const float*)d_in[1];
  const float* b_attn = (const float*)d_in[2];
  const float* w_proj = (const float*)d_in[3];
  const float* b_proj = (const float*)d_in[4];
  float* out = (float*)d_out;

  const int B = 4, S = 1024, E = 1024, H = 16, D = 64;
  const int M1 = B * S;      // 4096
  const int E3 = 3 * E;      // 3072
  const int Z  = B * H;      // 64
  const long long sBH = (long long)S * E3;   // batch-b stride inside QKV
  const long long SD  = (long long)S * D;
  const long long SS  = (long long)S * S;

  char* ws = (char*)d_ws;
  size_t off = 0;
  auto alloc = [&](size_t bytes) -> void* {
    void* p = ws + off;
    off = (off + bytes + 255) & ~(size_t)255;
    return p;
  };
  __hip_bfloat16* Xq   = (__hip_bfloat16*)alloc((size_t)M1 * E  * 2);
  __hip_bfloat16* Wtq  = (__hip_bfloat16*)alloc((size_t)E3 * E  * 2);
  float*          QKV  = (float*)         alloc((size_t)M1 * E3 * 4);
  __hip_bfloat16* Qq   = (__hip_bfloat16*)alloc((size_t)Z * SD * 2);
  __hip_bfloat16* Ktq  = (__hip_bfloat16*)alloc((size_t)Z * SD * 2);
  __hip_bfloat16* Vtq  = (__hip_bfloat16*)alloc((size_t)Z * SD * 2);
  float*          AW   = (float*)         alloc((size_t)Z * SS * 4);
  __hip_bfloat16* AWq  = (__hip_bfloat16*)alloc((size_t)Z * SS * 2);
  float*          CTX  = (float*)         alloc((size_t)M1 * E * 4);
  __hip_bfloat16* CTXq = (__hip_bfloat16*)alloc((size_t)M1 * E * 2);
  __hip_bfloat16* WPtq = (__hip_bfloat16*)alloc((size_t)E * E * 2);
  float* s1x  = (float*)alloc((size_t)E  * 4);
  float* s1w  = (float*)alloc((size_t)E3 * 4);
  float* s1q  = (float*)alloc((size_t)Z * D * 4);
  float* s1k  = (float*)alloc((size_t)Z * S * 4);
  float* s1v  = (float*)alloc((size_t)Z * D * 4);
  float* s1aw = (float*)alloc((size_t)Z * S * 4);
  float* s1c  = (float*)alloc((size_t)E * 4);
  float* s1p  = (float*)alloc((size_t)E * 4);

  dim3 blk(256);
  dim3 gblk(128);

  // ---- c_attn: dequant(x), dequant(w_attn)^T, GEMM + b_attn --------------
  colmax_pow2_kernel<<<dim3(E / 256, 1), blk, 0, stream>>>(hidden, s1x, M1, E, E, 1, 1, 0, 0);
  rowdq_kernel<<<dim3(M1, 1), blk, 0, stream>>>(hidden, s1x, Xq, E, E, 1, 1, 0, 0, 0, E, 0);
  colmax_pow2_kernel<<<dim3(E3 / 256, 1), blk, 0, stream>>>(w_attn, s1w, E, E3, E3, 1, 1, 0, 0);
  rowdq_kernel<<<dim3(E, 1), blk, 0, stream>>>(w_attn, s1w, Wtq, E3, E3, 1, 1, 0, 0, 0, E, 1);
  gemm_bf16_wmma_kernel<<<dim3(E3 / 64, M1 / 64, 1), gblk, 0, stream>>>(
      Xq, Wtq, QKV, b_attn, E, E, E, E3, 0, 0, 1, 0, 0);

  // ---- per-head dequant: Q [S,D], K^T stored as [S,D], V stored as [D,S] --
  colmax_pow2_kernel<<<dim3(1, Z), blk, 0, stream>>>(QKV, s1q, S, D, E3, 1, H, sBH, D);
  rowdq_kernel<<<dim3(S, Z), blk, 0, stream>>>(QKV, s1q, Qq, D, E3, 1, H, sBH, D, SD, D, 0);
  colmax_pow2_kernel<<<dim3(S / 256, Z), blk, 0, stream>>>(QKV + E, s1k, D, S, 1, E3, H, sBH, D);
  rowdq_kernel<<<dim3(D, Z), blk, 0, stream>>>(QKV + E, s1k, Ktq, S, 1, E3, H, sBH, D, SD, D, 1);
  colmax_pow2_kernel<<<dim3(1, Z), blk, 0, stream>>>(QKV + 2 * E, s1v, S, D, E3, 1, H, sBH, D);
  rowdq_kernel<<<dim3(S, Z), blk, 0, stream>>>(QKV + 2 * E, s1v, Vtq, D, E3, 1, H, sBH, D, SD, S, 1);

  // ---- scores = Qq @ Ktq^T, softmax, dequant ------------------------------
  gemm_bf16_wmma_kernel<<<dim3(S / 64, S / 64, Z), gblk, 0, stream>>>(
      Qq, Ktq, AW, nullptr, D, D, D, S, SD, SD, 1, SS, 0);
  softmax_causal_kernel<<<dim3(S, Z), blk, 0, stream>>>(AW, S, 0.125f);
  colmax_pow2_kernel<<<dim3(S / 256, Z), blk, 0, stream>>>(AW, s1aw, S, S, S, 1, 1, SS, 0);
  rowdq_kernel<<<dim3(S, Z), blk, 0, stream>>>(AW, s1aw, AWq, S, S, 1, 1, SS, 0, SS, S, 0);

  // ---- context = AWq @ Vtq^T, written with heads merged into [B,S,E] ------
  gemm_bf16_wmma_kernel<<<dim3(D / 64, S / 64, Z), gblk, 0, stream>>>(
      AWq, Vtq, CTX, nullptr, S, S, S, E, SS, SD, H, (long long)S * E, (long long)D);

  // ---- c_proj: dequant(ctx), dequant(w_proj)^T, GEMM + b_proj -> out ------
  colmax_pow2_kernel<<<dim3(E / 256, 1), blk, 0, stream>>>(CTX, s1c, M1, E, E, 1, 1, 0, 0);
  rowdq_kernel<<<dim3(M1, 1), blk, 0, stream>>>(CTX, s1c, CTXq, E, E, 1, 1, 0, 0, 0, E, 0);
  colmax_pow2_kernel<<<dim3(E / 256, 1), blk, 0, stream>>>(w_proj, s1p, E, E, E, 1, 1, 0, 0);
  rowdq_kernel<<<dim3(E, 1), blk, 0, stream>>>(w_proj, s1p, WPtq, E, E, 1, 1, 0, 0, 0, E, 1);
  gemm_bf16_wmma_kernel<<<dim3(E / 64, M1 / 64, 1), gblk, 0, stream>>>(
      CTXq, WPtq, out, b_proj, E, E, E, E, 0, 0, 1, 0, 0);
}
